// MixerGroupedTiedDifferentialAttention_50843822850200
// MI455X (gfx1250) — matrix operand order repaired and checked by reading, not compiled, not run on hardware
//
#include <hip/hip_runtime.h>
#include <hip/hip_bf16.h>

// ---------------------------------------------------------------------------
// Problem constants (from reference)
// ---------------------------------------------------------------------------
#define BB   2
#define TT   2048
#define DD   2048
#define HH   16
#define KVHH 4
#define HDD  128
#define NQKV 2560            // (H+KVH)*HD
#define NCAT 2624            // NQKV + D2 (k_rope proj)
#define MM   (BB*TT)         // 4096 flattened rows

typedef __bf16 bf16_t;
typedef __attribute__((ext_vector_type(16))) __bf16          bf16x16;
typedef __attribute__((ext_vector_type(8)))  float           f32x8;
typedef __attribute__((ext_vector_type(8)))  unsigned short  u16x8;
typedef __attribute__((ext_vector_type(16))) unsigned short  u16x16;

__device__ __forceinline__ unsigned short f2bf(float f) {
    unsigned int u = __builtin_bit_cast(unsigned int, f);
    unsigned int r = (u + 0x7FFFu + ((u >> 16) & 1u)) >> 16;  // RNE
    return (unsigned short)r;
}

__device__ __forceinline__ f32x8 wmma_bf16(bf16x16 a, bf16x16 b, f32x8 c) {
    // D = A(16x32) * B(32x16) + C  -> v_wmma_f32_16x16x32_bf16
    return __builtin_amdgcn_wmma_f32_16x16x32_bf16(
        false, a, false, b, (short)0, c, false, false);
}

// A fragment 16x32 bf16 (row-major src, stride ld elements).
// ISA layout: lanes 0-15 -> row=lane, K{0..7,16..23}; lanes 16-31 -> K{8..15,24..31}
__device__ __forceinline__ bf16x16 load_a16x32(const unsigned short* base, int ld, int lane) {
    const int hf = (lane >> 4) & 1;
    const int r  = lane & 15;
    const unsigned short* p = base + (size_t)r * ld + hf * 8;
    u16x8 lo = *(const u16x8*)(p);
    u16x8 hi = *(const u16x8*)(p + 16);
    u16x16 f;
#pragma unroll
    for (int i = 0; i < 8; ++i) { f[i] = lo[i]; f[i + 8] = hi[i]; }
    return __builtin_bit_cast(bf16x16, f);
}

// B fragment 32(K)x16(N) bf16 where memory holds B^T rows: src[n][k], stride ld.
// ISA layout: lane = column n; lanes 0-15 hold K{0..7,16..23}, lanes 16-31 hold K{8..15,24..31}
__device__ __forceinline__ bf16x16 load_b32x16(const unsigned short* base, int ld, int lane) {
    const int hf = (lane >> 4) & 1;
    const int n  = lane & 15;
    const unsigned short* p = base + (size_t)n * ld + hf * 8;
    u16x8 lo = *(const u16x8*)(p);
    u16x8 hi = *(const u16x8*)(p + 16);
    u16x16 f;
#pragma unroll
    for (int i = 0; i < 8; ++i) { f[i] = lo[i]; f[i + 8] = hi[i]; }
    return __builtin_bit_cast(bf16x16, f);
}

// ---------------------------------------------------------------------------
// Kernel 0a: x fp32 -> bf16
// ---------------------------------------------------------------------------
__global__ __launch_bounds__(256) void k_cvt_x(const float* __restrict__ x,
                                               unsigned short* __restrict__ xb) {
    size_t i = (size_t)blockIdx.x * 256 + threadIdx.x;
    if (i < (size_t)MM * DD) xb[i] = f2bf(x[i]);
}

// ---------------------------------------------------------------------------
// Kernel 0b: Wt[n][k] = concat(W_qkv, W_rope_k)[k][n]  (bf16, transposed)
// ---------------------------------------------------------------------------
__global__ __launch_bounds__(256) void k_cvt_w(const float* __restrict__ wqkv,
                                               const float* __restrict__ wrope,
                                               unsigned short* __restrict__ wt) {
    size_t i = (size_t)blockIdx.x * 256 + threadIdx.x;
    if (i >= (size_t)NCAT * DD) return;
    int n = (int)(i / DD);
    int k = (int)(i % DD);
    float v = (n < NQKV) ? wqkv[(size_t)k * NQKV + n]
                         : wrope[(size_t)k * 64 + (n - NQKV)];
    wt[i] = f2bf(v);
}

// ---------------------------------------------------------------------------
// Kernel 1: qkv_raw[m][n] = x @ Wcat  (bf16 WMMA, fp32 accum)
// wave tile = 16x64, K-loop step 32
// ---------------------------------------------------------------------------
__global__ __launch_bounds__(256) void k_gemm(const unsigned short* __restrict__ xb,
                                              const unsigned short* __restrict__ wt,
                                              float* __restrict__ outp) {
    const int lane = threadIdx.x & 31;
    const int wid  = threadIdx.x >> 5;
    const int w    = blockIdx.x * 8 + wid;
    const int NT   = NCAT / 64;   // 41 n-groups
    const int mt   = w / NT;
    const int ng   = w % NT;
    if (mt >= MM / 16) return;
    const int m0 = mt * 16, n0 = ng * 64;

    const f32x8 z = {0.f, 0.f, 0.f, 0.f, 0.f, 0.f, 0.f, 0.f};
    f32x8 c0 = z, c1 = z, c2 = z, c3 = z;

    const unsigned short* abase = xb + (size_t)m0 * DD;
    for (int k0 = 0; k0 < DD; k0 += 32) {
        __builtin_prefetch(abase + k0 + 128, 0, 1);   // global_prefetch_b8
        bf16x16 a  = load_a16x32(abase + k0, DD, lane);
        bf16x16 b0 = load_b32x16(wt + (size_t)(n0 +  0) * DD + k0, DD, lane);
        bf16x16 b1 = load_b32x16(wt + (size_t)(n0 + 16) * DD + k0, DD, lane);
        bf16x16 b2 = load_b32x16(wt + (size_t)(n0 + 32) * DD + k0, DD, lane);
        bf16x16 b3 = load_b32x16(wt + (size_t)(n0 + 48) * DD + k0, DD, lane);
        c0 = wmma_bf16(a, b0, c0);
        c1 = wmma_bf16(a, b1, c1);
        c2 = wmma_bf16(a, b2, c2);
        c3 = wmma_bf16(a, b3, c3);
    }
    const int hf = (lane >> 4) & 1, cc = lane & 15;
#pragma unroll
    for (int j = 0; j < 8; ++j) {
        int mrow = m0 + 8 * hf + j;
        float* orow = outp + (size_t)mrow * NCAT + n0 + cc;
        orow[0]  = c0[j];
        orow[16] = c1[j];
        orow[32] = c2[j];
        orow[48] = c3[j];
    }
}

// ---------------------------------------------------------------------------
// Kernel 2: RMSNorm + RoPE + scaling, emit bf16 q/k [B,H,T,128] and Vt [B,4,128,T]
// one 128-thread block per (b,t)
// ---------------------------------------------------------------------------
__global__ __launch_bounds__(128) void k_post(const float* __restrict__ qkvraw,
                                              const float* __restrict__ brope,
                                              const float* __restrict__ sscal,
                                              unsigned short* __restrict__ qb,
                                              unsigned short* __restrict__ kb,
                                              unsigned short* __restrict__ vt) {
    __shared__ float sraw[NCAT];
    __shared__ float srms[20];
    const int bt = blockIdx.x;
    const int b  = bt / TT;
    const int t  = bt % TT;
    const int tid = threadIdx.x;

    const float* src = qkvraw + (size_t)bt * NCAT;
    for (int i = tid; i < NCAT; i += 128) sraw[i] = src[i];
    __syncthreads();

    const int wid = tid >> 5, lane = tid & 31;
    for (int h = wid; h < 20; h += 4) {
        float s = 0.f;
#pragma unroll
        for (int k = 0; k < 4; ++k) {
            float v = sraw[h * 128 + lane + 32 * k];
            s += v * v;
        }
        s += __shfl_xor(s, 1, 32);
        s += __shfl_xor(s, 2, 32);
        s += __shfl_xor(s, 4, 32);
        s += __shfl_xor(s, 8, 32);
        s += __shfl_xor(s, 16, 32);
        if (lane == 0) srms[h] = rsqrtf(s * (1.f / 128.f) + 1e-6f);
    }
    __syncthreads();

    const float tf     = (float)t;
    const float logpos = logf((float)(t + 1));
    const float qmul   = logpos * 0.08838834764831845f;  // log(pos)/sqrt(128)

    // q heads
    for (int i = tid; i < HH * 128; i += 128) {
        int h = i >> 7, d = i & 127;
        float rms = srms[h];
        float val;
        if (d < 64) {
            val = sraw[h * 128 + d] * rms;
        } else {
            int r  = d - 64;
            int fr = (r < 32) ? r : r - 32;
            float inv = powf(10000.f, -(float)fr * (1.f / 32.f));
            float cs = cosf(tf * inv), sn = sinf(tf * inv);
            float x1 = sraw[h * 128 + 64 + fr] * rms;
            float x2 = sraw[h * 128 + 96 + fr] * rms;
            val = (r < 32) ? (x1 * cs + x2 * sn) : (x2 * cs - x1 * sn);
        }
        val *= sscal[h] * qmul;
        qb[(((size_t)b * HH + h) * TT + t) * 128 + d] = f2bf(val);
    }
    // k heads: tied part from kv head h/4, rope part from raw k_rope proj (+bias)
    for (int i = tid; i < HH * 128; i += 128) {
        int h = i >> 7, d = i & 127;
        float val;
        if (d < 64) {
            int g = 16 + (h >> 2);
            val = sraw[g * 128 + d] * srms[g];
        } else {
            int r  = d - 64;
            int fr = (r < 32) ? r : r - 32;
            float inv = powf(10000.f, -(float)fr * (1.f / 32.f));
            float cs = cosf(tf * inv), sn = sinf(tf * inv);
            float x1 = sraw[NQKV + fr]      + brope[fr];
            float x2 = sraw[NQKV + 32 + fr] + brope[32 + fr];
            val = (r < 32) ? (x1 * cs + x2 * sn) : (x2 * cs - x1 * sn);
        }
        kb[(((size_t)b * HH + h) * TT + t) * 128 + d] = f2bf(val);
    }
    // v: full normalized kv head (tied ++ hidden) transposed -> Vt[b][g][d][t]
    for (int i = tid; i < KVHH * 128; i += 128) {
        int g = i >> 7, d = i & 127;
        float val = sraw[(16 + g) * 128 + d] * srms[16 + g];
        vt[(((size_t)b * KVHH + g) * 128 + d) * TT + t] = f2bf(val);
    }
}

// ---------------------------------------------------------------------------
// Kernel 3: causal flash attention, one wave per (b, h, 16-row q tile)
// ---------------------------------------------------------------------------
__global__ __launch_bounds__(128) void k_attn(const unsigned short* __restrict__ qb,
                                              const unsigned short* __restrict__ kbuf,
                                              const unsigned short* __restrict__ vtb,
                                              float* __restrict__ obuf) {
    __shared__ unsigned short sP[4][16 * 32];   // per-wave P staging (C->A reshape)

    const int lane = threadIdx.x & 31;
    const int wid  = threadIdx.x >> 5;
    const int w    = blockIdx.x * 4 + wid;
    const int tilesPerHead = TT / 16;  // 128
    const int b   = w / (HH * tilesPerHead);
    const int rem = w % (HH * tilesPerHead);
    const int h   = rem / tilesPerHead;
    const int mt  = rem % tilesPerHead;
    const int m0  = mt * 16;

    const unsigned short* qp = qb   + (((size_t)b * HH   + h)        * TT) * 128;
    const unsigned short* kp = kbuf + (((size_t)b * HH   + h)        * TT) * 128;
    const unsigned short* vp = vtb  + (((size_t)b * KVHH + (h >> 2)) * 128) * TT;

    bf16x16 qf[4];
#pragma unroll
    for (int kc = 0; kc < 4; ++kc)
        qf[kc] = load_a16x32(qp + (size_t)m0 * 128 + kc * 32, 128, lane);

    const f32x8 z = {0.f, 0.f, 0.f, 0.f, 0.f, 0.f, 0.f, 0.f};
    f32x8 acc[8];
    float m_run[8], l_run[8];
#pragma unroll
    for (int n = 0; n < 8; ++n) acc[n] = z;
#pragma unroll
    for (int j = 0; j < 8; ++j) { m_run[j] = -3.0e38f; l_run[j] = 0.f; }

    const int hf = (lane >> 4) & 1, cc = lane & 15;

    for (int kbk = 0; kbk < m0 + 16; kbk += 32) {
        // ---- S = Q * K^T for 32 keys (two 16-key subtiles) ----
        f32x8 s0 = z, s1 = z;
#pragma unroll
        for (int kc = 0; kc < 4; ++kc) {
            bf16x16 b0 = load_b32x16(kp + (size_t)kbk * 128 + kc * 32, 128, lane);
            s0 = wmma_bf16(qf[kc], b0, s0);
        }
#pragma unroll
        for (int kc = 0; kc < 4; ++kc) {
            bf16x16 b1 = load_b32x16(kp + (size_t)(kbk + 16) * 128 + kc * 32, 128, lane);
            s1 = wmma_bf16(qf[kc], b1, s1);
        }

        // ---- causal mask + online softmax (row stats across 16-lane halves) ----
        float mnew[8], alpha[8];
#pragma unroll
        for (int j = 0; j < 8; ++j) {
            int row = m0 + 8 * hf + j;
            float a0 = ((kbk + cc)      > row) ? -1e30f : s0[j];
            float a1 = ((kbk + 16 + cc) > row) ? -1e30f : s1[j];
            s0[j] = a0; s1[j] = a1;
            float r = fmaxf(a0, a1);
            r = fmaxf(r, __shfl_xor(r, 1, 32));
            r = fmaxf(r, __shfl_xor(r, 2, 32));
            r = fmaxf(r, __shfl_xor(r, 4, 32));
            r = fmaxf(r, __shfl_xor(r, 8, 32));
            float mn = fmaxf(m_run[j], r);
            alpha[j] = expf(m_run[j] - mn);
            mnew[j]  = mn;
            m_run[j] = mn;
        }
#pragma unroll
        for (int j = 0; j < 8; ++j) {
            float p0 = expf(s0[j] - mnew[j]);
            float p1 = expf(s1[j] - mnew[j]);
            int row = 8 * hf + j;
            sP[wid][row * 32 + cc]      = f2bf(p0);
            sP[wid][row * 32 + 16 + cc] = f2bf(p1);
            float rs = p0 + p1;
            rs += __shfl_xor(rs, 1, 32);
            rs += __shfl_xor(rs, 2, 32);
            rs += __shfl_xor(rs, 4, 32);
            rs += __shfl_xor(rs, 8, 32);
            l_run[j] = l_run[j] * alpha[j] + rs;
#pragma unroll
            for (int n = 0; n < 8; ++n) acc[n][j] *= alpha[j];
        }

        // ---- P (C layout) -> A fragment via LDS (same-wave DS ops are ordered) ----
        const unsigned short* pb = &sP[wid][(lane & 15) * 32 + hf * 8];
        u16x8 plo = *(const u16x8*)(pb);
        u16x8 phi = *(const u16x8*)(pb + 16);
        u16x16 pfv;
#pragma unroll
        for (int i = 0; i < 8; ++i) { pfv[i] = plo[i]; pfv[i + 8] = phi[i]; }
        bf16x16 pa = __builtin_bit_cast(bf16x16, pfv);

        // ---- O += P * V  (8 dim-tiles of 16) ----
#pragma unroll
        for (int n = 0; n < 8; ++n) {
            bf16x16 vb = load_b32x16(vp + (size_t)(n * 16) * TT + kbk, TT, lane);
            acc[n] = wmma_bf16(pa, vb, acc[n]);
        }
    }

    float* op = obuf + (((size_t)b * HH + h) * TT) * 128;
#pragma unroll
    for (int j = 0; j < 8; ++j) {
        float inv = 1.f / l_run[j];
        int row = m0 + 8 * hf + j;
#pragma unroll
        for (int n = 0; n < 8; ++n)
            op[(size_t)row * 128 + n * 16 + cc] = acc[n][j] * inv;
    }
}

// ---------------------------------------------------------------------------
// Kernel 4: y = o1 - lambda*o2, write both identical 128-halves of the 256 dim
// ---------------------------------------------------------------------------
__global__ __launch_bounds__(256) void k_combine(const float* __restrict__ obuf,
                                                 const float* __restrict__ lq1,
                                                 const float* __restrict__ lk1,
                                                 const float* __restrict__ lq2,
                                                 const float* __restrict__ lk2,
                                                 float* __restrict__ outp) {
    size_t idx = (size_t)blockIdx.x * 256 + threadIdx.x;
    const size_t total = (size_t)BB * TT * 8 * 128;
    if (idx >= total) return;

    float s1 = 0.f, s2 = 0.f;
#pragma unroll
    for (int i = 0; i < 64; ++i) {
        s1 += lq1[i] * lk1[i];
        s2 += lq2[i] * lk2[i];
    }
    float lam = expf(s1) - expf(s2) + (0.8f - 0.6f);  // LAMBDA_INIT = 0.2

    int d = (int)(idx & 127);
    size_t r  = idx >> 7;
    int p     = (int)(r & 7);
    size_t bt = r >> 3;
    int t = (int)(bt % TT);
    int b = (int)(bt / TT);

    float o1 = obuf[(((size_t)b * HH + 2 * p)     * TT + t) * 128 + d];
    float o2 = obuf[(((size_t)b * HH + 2 * p + 1) * TT + t) * 128 + d];
    float y = o1 - lam * o2;
    float* yp = outp + (bt * 8 + p) * 256 + d;
    yp[0]   = y;
    yp[128] = y;
}

// ---------------------------------------------------------------------------
// Host launcher
// ---------------------------------------------------------------------------
extern "C" void kernel_launch(void* const* d_in, const int* in_sizes, int n_in,
                              void* d_out, int out_size, void* d_ws, size_t ws_size,
                              hipStream_t stream) {
    const float* x     = (const float*)d_in[0];
    const float* wqkv  = (const float*)d_in[1];
    const float* wrope = (const float*)d_in[2];
    const float* brope = (const float*)d_in[3];
    const float* sscal = (const float*)d_in[4];
    const float* lq1   = (const float*)d_in[5];
    const float* lk1   = (const float*)d_in[6];
    const float* lq2   = (const float*)d_in[7];
    const float* lk2   = (const float*)d_in[8];
    float* outp = (float*)d_out;

    char* ws = (char*)d_ws;
    // workspace layout (bytes)
    const size_t off_xb  = 0;
    const size_t off_wt  = off_xb  + (size_t)MM * DD * 2;          // x bf16
    const size_t off_raw = off_wt  + (size_t)NCAT * DD * 2;        // Wt bf16
    const size_t off_q   = off_raw + (size_t)MM * NCAT * 4;        // qkv raw f32
    const size_t off_k   = off_q   + (size_t)BB * HH * TT * 128 * 2;
    const size_t off_v   = off_k   + (size_t)BB * HH * TT * 128 * 2;
    // attention output aliases qkv_raw region (consumed before it is written)
    const size_t off_o   = off_raw;

    unsigned short* xb  = (unsigned short*)(ws + off_xb);
    unsigned short* wt  = (unsigned short*)(ws + off_wt);
    float*          raw = (float*)(ws + off_raw);
    unsigned short* qbb = (unsigned short*)(ws + off_q);
    unsigned short* kbb = (unsigned short*)(ws + off_k);
    unsigned short* vtb = (unsigned short*)(ws + off_v);
    float*          obf = (float*)(ws + off_o);

    {   // convert x
        size_t n = (size_t)MM * DD;
        k_cvt_x<<<(unsigned)((n + 255) / 256), 256, 0, stream>>>(x, xb);
    }
    {   // convert + transpose weights
        size_t n = (size_t)NCAT * DD;
        k_cvt_w<<<(unsigned)((n + 255) / 256), 256, 0, stream>>>(wqkv, wrope, wt);
    }
    {   // GEMM: (M/16)*(NCAT/64) waves, 8 waves per block
        unsigned waves = (MM / 16) * (NCAT / 64);  // 256*41 = 10496
        k_gemm<<<waves / 8, 256, 0, stream>>>(xb, wt, raw);
    }
    {   // postprocess: one block per (b,t)
        k_post<<<MM, 128, 0, stream>>>(raw, brope, sscal, qbb, kbb, vtb);
    }
    {   // attention: B*H*(T/16) waves, 4 per block
        unsigned waves = BB * HH * (TT / 16);      // 4096
        k_attn<<<waves / 4, 128, 0, stream>>>(qbb, kbb, vtb, obf);
    }
    {   // combine
        size_t n = (size_t)BB * TT * 8 * 128;
        k_combine<<<(unsigned)((n + 255) / 256), 256, 0, stream>>>(obf, lq1, lk1, lq2, lk2, outp);
    }
    (void)in_sizes; (void)n_in; (void)out_size; (void)ws_size;
}
